// GTLayer_46067819217157
// MI455X (gfx1250) — compile-verified, matrix operand order checked by reading, not run on hardware
//
#include <hip/hip_runtime.h>
#include <hip/hip_bf16.h>
#include <math.h>

// Problem constants (match reference)
#define NN      50000
#define EDGES   800000
#define NUM_A   4
#define CHAN    2
#define D_IN    128
#define D_OUT   64

#define ROW_TILES (NN / 16)          // 3125
#define COL_TILES (D_OUT / 16)       // 4
#define WPITCH    (D_IN + 2)         // 130 bf16 -> 65-dword row stride, bank-conflict free

typedef __attribute__((ext_vector_type(16))) __bf16 v16bf;
typedef __attribute__((ext_vector_type(2)))  __bf16 bf16x2;
typedef __attribute__((ext_vector_type(8)))  float  v8f;

// ---------------------------------------------------------------------------
// Kernel 0: zero the output accumulator
// ---------------------------------------------------------------------------
__global__ void zero_kernel(float* __restrict__ out, int n) {
    int i = blockIdx.x * blockDim.x + threadIdx.x;
    int stride = gridDim.x * blockDim.x;
    for (; i < n; i += stride) out[i] = 0.0f;
}

// ---------------------------------------------------------------------------
// Kernel 1: scores = softmax(mean_d att_weight[c,d,a], axis=a)  -> ws[0..7]
// att_weight layout: [C, D_IN, NUM_A] row-major
// ---------------------------------------------------------------------------
__global__ void scores_kernel(const float* __restrict__ att_weight,
                              float* __restrict__ score) {
    __shared__ float s[CHAN * NUM_A];
    int t = threadIdx.x;
    if (t < CHAN * NUM_A) {
        int c = t >> 2, a = t & 3;
        float sum = 0.0f;
        for (int d = 0; d < D_IN; ++d)
            sum += att_weight[((size_t)c * D_IN + d) * NUM_A + a];
        s[t] = sum * (1.0f / D_IN);
    }
    __syncthreads();
    if (t < CHAN) {
        float m = s[t * NUM_A];
        for (int a = 1; a < NUM_A; ++a) m = fmaxf(m, s[t * NUM_A + a]);
        float e[NUM_A], tot = 0.0f;
        for (int a = 0; a < NUM_A; ++a) { e[a] = __expf(s[t * NUM_A + a] - m); tot += e[a]; }
        float inv = 1.0f / tot;
        for (int a = 0; a < NUM_A; ++a) score[t * NUM_A + a] = e[a] * inv;
    }
}

// ---------------------------------------------------------------------------
// Kernel 2: Hc[c] = H @ weight[c]   via v_wmma_f32_16x16x32_bf16
//   Grid: (ceil(ROW_TILES/2), CHAN). Block: 256 thr = 8 waves
//         = 2 row-tiles x 4 col-tiles, one 16x16 tile per wave.
//   weight[c] (128x64 f32) is staged ONCE per block into LDS, transposed to
//   [o][k] bf16 with pitch 130 (65-dword row stride => conflict-free column
//   reads). B fragments then come from LDS as packed bf16 pairs; A fragments
//   from coalesced global b128 loads + cvt_pk_bf16.
// ---------------------------------------------------------------------------
__global__ void __launch_bounds__(256)
gemm_wmma_kernel(const float* __restrict__ H,
                 const float* __restrict__ W,
                 float* __restrict__ Hc) {
    __shared__ __bf16 Wt[D_OUT * WPITCH];   // 64 * 130 * 2B = 16.25 KB

    const int tid  = threadIdx.x;
    const int lane = tid & 31;
    const int wave = tid >> 5;
    const int c    = blockIdx.y;

    // --- stage weight[c] into LDS, transposed + converted to bf16 ----------
    const float* __restrict__ Wg = W + (size_t)c * D_IN * D_OUT;
    #pragma unroll
    for (int i = tid; i < D_IN * D_OUT; i += 256) {
        const int k = i >> 6;        // 0..127
        const int o = i & 63;        // 0..63
        Wt[o * WPITCH + k] = (__bf16)Wg[i];   // coalesced global read
    }
    __syncthreads();

    // --- wave -> tile mapping (wave-uniform guard AFTER the barrier) -------
    const int rowTile = blockIdx.x * 2 + (wave >> 2);
    if (rowTile >= ROW_TILES) return;        // uniform across the wave
    const int colBase = (wave & 3) * 16;
    const int rowBase = rowTile * 16;

    const int g  = lane >> 4;     // lane group (0/1) selects K sub-range
    const int ln = lane & 15;     // M for A-frag, N for B-frag

    const float*  __restrict__ hrow = H + (size_t)(rowBase + ln) * D_IN;
    const __bf16* __restrict__ wrow = Wt + (colBase + ln) * WPITCH;

    v8f acc = {};
    #pragma unroll
    for (int kk = 0; kk < D_IN; kk += 32) {
        v16bf a, b;
        // ISA 7.12.2 16-bit A 16x32: VGPR j<4 -> K = 8g + 2j ; j>=4 -> K = 16 + 8g + 2(j-4)
        // B 32x16 mirrored with N = lane&15; pairs read as packed bf16x2 from LDS.
        #pragma unroll
        for (int j = 0; j < 8; ++j) {
            const int k = (j < 4) ? (kk + g * 8 + 2 * j)
                                  : (kk + 16 + g * 8 + 2 * (j - 4));
            a[2 * j]     = (__bf16)hrow[k];
            a[2 * j + 1] = (__bf16)hrow[k + 1];
            const bf16x2 wp = *(const bf16x2*)(wrow + k);   // 4B-aligned (k even, pitch even)
            b[2 * j]     = wp.x;
            b[2 * j + 1] = wp.y;
        }
        acc = __builtin_amdgcn_wmma_f32_16x16x32_bf16(
                  /*neg_a=*/false, a, /*neg_b=*/false, b,
                  /*c_mod=*/(short)0, acc, /*reuse_a=*/false, /*reuse_b=*/false);
    }

    // D layout: VGPR j -> row = rowBase + 8g + j, col = colBase + (lane&15)
    float* __restrict__ outBase = Hc + (size_t)c * NN * D_OUT;
    #pragma unroll
    for (int j = 0; j < 8; ++j)
        outBase[(size_t)(rowBase + g * 8 + j) * D_OUT + colBase + ln] = acc[j];
}

// ---------------------------------------------------------------------------
// Kernel 3: fused edge scatter.
//   For flat edge eid in [0, NUM_A*EDGES):
//     a = eid / EDGES ; r = rows[eid] ; cl = cols[eid] ; v = vals[eid]
//     out[r, :] += 0.5*score[0,a]*v * Hc0[cl, :] + 0.5*score[1,a]*v * Hc1[cl, :]
//   64 lanes per edge -> fully coalesced 256B gathers (Hc is L2-resident,
//   25.6 MB << 192 MB) and 256B atomic-add bursts. 4 edges per 256-thr block.
// ---------------------------------------------------------------------------
__global__ void __launch_bounds__(256)
scatter_kernel(const float* __restrict__ vals,
               const int*   __restrict__ rows,
               const int*   __restrict__ cols,
               const float* __restrict__ Hc,
               const float* __restrict__ score,
               float*       __restrict__ out) {
    const long long eid = (long long)blockIdx.x * 4 + (threadIdx.x >> 6);
    const int o = threadIdx.x & 63;
    if (eid >= (long long)NUM_A * EDGES) return;

    const int   a  = (int)(eid / EDGES);
    const int   r  = rows[eid];
    const int   cl = cols[eid];
    const float v  = vals[eid];

    const float w0 = 0.5f * score[a] * v;           // channel 0 (mean folded in)
    const float w1 = 0.5f * score[NUM_A + a] * v;   // channel 1

    const float h0 = Hc[(size_t)cl * D_OUT + o];
    const float h1 = Hc[(size_t)NN * D_OUT + (size_t)cl * D_OUT + o];

    atomicAdd(&out[(size_t)r * D_OUT + o], w0 * h0 + w1 * h1);
}

// ---------------------------------------------------------------------------
// Launch
//   d_in order (setup_inputs): H, vals, weight, att_weight, rows, cols
//   d_ws layout: [0,256) scores (8 floats) ; [256, 256 + 25.6MB) Hc
// ---------------------------------------------------------------------------
extern "C" void kernel_launch(void* const* d_in, const int* in_sizes, int n_in,
                              void* d_out, int out_size, void* d_ws, size_t ws_size,
                              hipStream_t stream) {
    const float* H          = (const float*)d_in[0];
    const float* vals       = (const float*)d_in[1];
    const float* weight     = (const float*)d_in[2];
    const float* att_weight = (const float*)d_in[3];
    const int*   rows       = (const int*)d_in[4];
    const int*   cols       = (const int*)d_in[5];
    float*       out        = (float*)d_out;

    float* score = (float*)d_ws;
    float* Hc    = (float*)((char*)d_ws + 256);

    // 1) zero output accumulator
    zero_kernel<<<2048, 256, 0, stream>>>(out, NN * D_OUT);

    // 2) attention scores (8 scalars)
    scores_kernel<<<1, 32, 0, stream>>>(att_weight, score);

    // 3) Hc = H @ W[c]  (WMMA bf16 -> f32), weight staged in LDS per block
    dim3 ggrid((ROW_TILES + 1) / 2, CHAN);               // (1563, 2)
    gemm_wmma_kernel<<<ggrid, 256, 0, stream>>>(H, weight, Hc);

    // 4) fused SpMM scatter (4 edges per block)
    const long long nedges = (long long)NUM_A * EDGES;   // 3.2M
    const int sblocks = (int)((nedges + 3) / 4);         // 800000
    scatter_kernel<<<sblocks, 256, 0, stream>>>(vals, rows, cols, Hc, score, out);
}
